// MultiHeadAttention_57432302682608
// MI455X (gfx1250) — matrix operand order repaired and checked by reading, not compiled, not run on hardware
//
#include <hip/hip_runtime.h>
#include <stdint.h>

// ---------------------------------------------------------------------------
// MHA block for MI455X (gfx1250, wave32, WMMA + TDM).
// Pipeline: fp32->bf16 convert, QKV GEMM (WMMA bf16, TDM-fed double-buffered
// LDS), causal flash attention (WMMA bf16 for QK^T and PV, f32 online
// softmax, L2 prefetch of next tiles), output proj GEMM (f32 out).
// ---------------------------------------------------------------------------

typedef __bf16 bf16_t;
typedef __attribute__((ext_vector_type(16))) __bf16 bf16x16;
typedef __attribute__((ext_vector_type(8)))  float  f32x8;
typedef __attribute__((ext_vector_type(4)))  unsigned int u32x4;
typedef __attribute__((ext_vector_type(8)))  int  i32x8;
typedef __attribute__((ext_vector_type(4)))  int  i32x4;

#define B_DIM 4
#define T_DIM 2048
#define C_DIM 1024
#define H_DIM 16
#define HD_DIM 64

__device__ __forceinline__ bf16_t f32_to_bf16(float f) {
  union { float f; uint32_t u; } in; in.f = f;
  uint32_t u = in.u;
  uint32_t r = (u + 0x7FFFu + ((u >> 16) & 1u)) >> 16;  // round-to-nearest-even
  union { uint16_t s; bf16_t b; } out; out.s = (uint16_t)r;
  return out.b;
}

union FragU { bf16x16 v; uint4 q[2]; };

// LDS byte offset of a __shared__ object (generic -> AS(3) -> int).
__device__ __forceinline__ uint32_t lds_offset(const void* p) {
  return (uint32_t)(uintptr_t)(__attribute__((address_space(3))) const void*)p;
}

// ---------------------------------------------------------------------------
// Tensor Data Mover: 2D bf16 tile load (global -> LDS), row-major packed.
// D# per CDNA5 ISA ch.8: group0 {count=1, lds_addr, global_addr, type=2},
// group1 {data_size=2B, tensor dims, tile dims, dim0 stride}. Tensor dims are
// set equal to the tile dims (tiles never cross our matrix bounds), so the
// OOB clip path is never exercised. TENSORcnt tracks completion per wave.
// This toolchain exposes the 6-arg builtin:
//   (u32x4 g0, i32x8 g1, i32x4 g2, i32x4 g3, i32x8 ext, i32 cpol)
// ---------------------------------------------------------------------------
__device__ __forceinline__ void tdm_load_tile_bf16(uint32_t lds_byte,
                                                   const void* gaddr,
                                                   uint32_t tile_d0,   // elems, contiguous dim
                                                   uint32_t tile_d1,   // rows
                                                   uint64_t stride_d0) // elems between rows
{
  uint64_t ga = (uint64_t)(uintptr_t)gaddr;
  u32x4 g0;
  g0[0] = 1u;                                            // count=1, user D#
  g0[1] = lds_byte;                                      // lds_addr (bits 63:32)
  g0[2] = (uint32_t)(ga & 0xFFFFFFFFu);                  // global_addr[31:0]
  g0[3] = (uint32_t)((ga >> 32) & 0x01FFFFFFu) | (2u << 30); // addr[56:32], type=2
  i32x8 g1;
  g1[0] = (int)(1u << 16);                               // data_size=1 (2 bytes)
  g1[1] = (int)((tile_d0 & 0xFFFFu) << 16);              // tensor_dim0[15:0] @ bits63:48
  g1[2] = (int)(((tile_d0 >> 16) & 0xFFFFu) | ((tile_d1 & 0xFFFFu) << 16)); // td0 hi, td1 lo
  g1[3] = (int)(((tile_d1 >> 16) & 0xFFFFu) | ((tile_d0 & 0xFFFFu) << 16)); // td1 hi, tile_dim0
  g1[4] = (int)(tile_d1 & 0xFFFFu);                      // tile_dim1, tile_dim2=0
  g1[5] = (int)(uint32_t)(stride_d0 & 0xFFFFFFFFu);      // tensor_dim0_stride[31:0]
  g1[6] = (int)(uint32_t)((stride_d0 >> 32) & 0xFFFFu);  // stride[47:32], dim1_stride lo = 0
  g1[7] = 0;
  i32x4 z4 = {0, 0, 0, 0};
  i32x8 z8 = {0, 0, 0, 0, 0, 0, 0, 0};
  __builtin_amdgcn_tensor_load_to_lds(g0, g1, z4, z4, z8, 0);
}

// ---------------------------------------------------------------------------
// fp32 -> bf16 elementwise convert
// ---------------------------------------------------------------------------
__global__ void convert_f32_bf16(const float* __restrict__ src,
                                 bf16_t* __restrict__ dst, int n) {
  int i = blockIdx.x * blockDim.x + threadIdx.x;
  int stride = gridDim.x * blockDim.x;
  for (; i < n; i += stride) dst[i] = f32_to_bf16(src[i]);
}

// ---------------------------------------------------------------------------
// Tiled GEMM: C[M,N] = A[M,K] * B[K,N], A/B bf16 row-major.
// Block tile 128x128, BK=32, 256 threads = 8 waves, wave tile 64x32.
// Double-buffered LDS: A tiles DMA'd by TDM (each wave moves its own 16-row
// slice, waits its own TENSORcnt), B tiles via global->reg->LDS transpose
// scatter so B-fragments are two contiguous b128 LDS loads per WMMA.
// A fragment (16-bit A 16x32): lane<16: row=lane, K 0..15;
// lane>=16: row=lane-16, K 16..31.
// B fragment (16-bit B 32x16): lane holds col=lane%16, K=(lane/16)*16..+15.
// ---------------------------------------------------------------------------
template <bool BF16OUT>
__global__ __launch_bounds__(256) void gemm_wmma(const bf16_t* __restrict__ A,
                                                 const bf16_t* __restrict__ Bm,
                                                 void* __restrict__ Cout,
                                                 int M, int N, int K) {
  __shared__ bf16_t As[2][128][32];    // 2 x 8 KB, TDM destination
  __shared__ bf16_t Bts[2][128][32];   // 2 x 8 KB, B transposed: Bts[.][n][k]

  const int tid  = threadIdx.x;
  const int lane = tid & 31;
  const int wave = tid >> 5;
  const int r = lane & 15;
  const int g = lane >> 4;
  const int mBase = blockIdx.y * 128;
  const int nBase = blockIdx.x * 128;
  const int wm = wave & 1;             // 2 waves along M
  const int wn = wave >> 1;            // 4 waves along N

  f32x8 zed = {};
  f32x8 acc[4][2];
#pragma unroll
  for (int mi = 0; mi < 4; ++mi)
#pragma unroll
    for (int ni = 0; ni < 2; ++ni) acc[mi][ni] = zed;

  // ---- prologue: stage tile 0 into buffer 0 ----
  tdm_load_tile_bf16(lds_offset(&As[0][wave * 16][0]),
                     &A[(size_t)(mBase + wave * 16) * K], 32, 16, (uint64_t)K);
  {
#pragma unroll
    for (int it = 0; it < 2; ++it) {
      int u = tid + it * 256;
      int krow = u >> 4;               // 16 uint4 per 128-wide row
      int n = (u & 15) << 3;
      union { uint4 q; bf16_t h[8]; } t;
      t.q = *(const uint4*)&Bm[(size_t)krow * N + nBase + n];
#pragma unroll
      for (int j = 0; j < 8; ++j) Bts[0][n + j][krow] = t.h[j];
    }
  }
  __builtin_amdgcn_s_wait_tensorcnt(0);
  __syncthreads();

  for (int k0 = 0; k0 < K; k0 += 32) {
    const int cur = (k0 >> 5) & 1;
    const int nxt = cur ^ 1;
    const bool more = (k0 + 32) < K;

    // Kick off next tile: TDM for A, global loads for B (held in registers).
    union { uint4 q; bf16_t h[8]; } bst[2];
    if (more) {
      tdm_load_tile_bf16(lds_offset(&As[nxt][wave * 16][0]),
                         &A[(size_t)(mBase + wave * 16) * K + k0 + 32],
                         32, 16, (uint64_t)K);
#pragma unroll
      for (int it = 0; it < 2; ++it) {
        int u = tid + it * 256;
        int krow = u >> 4;
        int n = (u & 15) << 3;
        bst[it].q = *(const uint4*)&Bm[(size_t)(k0 + 32 + krow) * N + nBase + n];
      }
    }

    // Compute on current buffers.
    FragU af[4], bf[2];
#pragma unroll
    for (int mi = 0; mi < 4; ++mi) {
      int row = wm * 64 + mi * 16 + r;
      af[mi].q[0] = *(const uint4*)&As[cur][row][g * 16];
      af[mi].q[1] = *(const uint4*)&As[cur][row][g * 16 + 8];
    }
#pragma unroll
    for (int ni = 0; ni < 2; ++ni) {
      int col = wn * 32 + ni * 16 + r;
      bf[ni].q[0] = *(const uint4*)&Bts[cur][col][g * 16];
      bf[ni].q[1] = *(const uint4*)&Bts[cur][col][g * 16 + 8];
    }
#pragma unroll
    for (int mi = 0; mi < 4; ++mi)
#pragma unroll
      for (int ni = 0; ni < 2; ++ni)
        acc[mi][ni] = __builtin_amdgcn_wmma_f32_16x16x32_bf16(
            false, af[mi].v, false, bf[ni].v, (short)0, acc[mi][ni], false, false);

    // Publish next buffer: scatter B, drain this wave's TDM, barrier.
    if (more) {
#pragma unroll
      for (int it = 0; it < 2; ++it) {
        int u = tid + it * 256;
        int krow = u >> 4;
        int n = (u & 15) << 3;
#pragma unroll
        for (int j = 0; j < 8; ++j) Bts[nxt][n + j][krow] = bst[it].h[j];
      }
      __builtin_amdgcn_s_wait_tensorcnt(0);
    }
    __syncthreads();
  }

  // Epilogue: C/D layout — VGPR i of lane: row = i + 8*(lane/16), col = lane%16.
#pragma unroll
  for (int mi = 0; mi < 4; ++mi)
#pragma unroll
    for (int ni = 0; ni < 2; ++ni)
#pragma unroll
      for (int i = 0; i < 8; ++i) {
        int grow = mBase + wm * 64 + mi * 16 + i + g * 8;
        int gcol = nBase + wn * 32 + ni * 16 + r;
        if (BF16OUT)
          ((bf16_t*)Cout)[(size_t)grow * N + gcol] = f32_to_bf16(acc[mi][ni][i]);
        else
          ((float*)Cout)[(size_t)grow * N + gcol] = acc[mi][ni][i];
      }
  (void)M;
}

// ---------------------------------------------------------------------------
// Causal flash attention. Grid: (T/128, B*H). 256 threads = 8 waves; wave w
// owns query rows [qblock*128 + w*16, +16). Online softmax in f32 registers.
// K fragments for S = Q*K^T come straight from global (K row-major matches the
// B-fragment layout). V is staged transposed in LDS; P goes accumulator-layout
// -> LDS -> A-fragment layout for the PV WMMA. Next key tile is prefetched.
// ---------------------------------------------------------------------------
__global__ __launch_bounds__(256) void flash_attn(const bf16_t* __restrict__ qkv,
                                                  bf16_t* __restrict__ out) {
  __shared__ bf16_t Vt[64][128];       // 16 KB, [d][key]
  __shared__ bf16_t Ps[8][16][128];    // 32 KB, per-wave P strip

  const int tid  = threadIdx.x;
  const int lane = tid & 31;
  const int wave = tid >> 5;
  const int r = lane & 15;
  const int g = lane >> 4;
  const int qblock = blockIdx.x;
  const int bh = blockIdx.y;
  const int bb = bh >> 4;              // / H
  const int hh = bh & 15;              // % H

  const int RS = 3 * C_DIM;            // qkv row stride in elements
  const size_t baseOff = (size_t)bb * T_DIM * RS + (size_t)hh * HD_DIM;
  const bf16_t* Qp = qkv + baseOff;
  const bf16_t* Kp = qkv + baseOff + C_DIM;
  const bf16_t* Vp = qkv + baseOff + 2 * C_DIM;
  const int q0 = qblock * 128 + wave * 16;

  // Q fragments for the two K-dim steps (d = 0..31, 32..63); held in registers.
  FragU qa[2];
#pragma unroll
  for (int kf = 0; kf < 2; ++kf) {
    int row = q0 + r;
    int kb = kf * 32 + g * 16;
    qa[kf].q[0] = *(const uint4*)&Qp[(size_t)row * RS + kb];
    qa[kf].q[1] = *(const uint4*)&Qp[(size_t)row * RS + kb + 8];
  }

  f32x8 zed = {};
  f32x8 o[4];
#pragma unroll
  for (int vt = 0; vt < 4; ++vt) o[vt] = zed;
  float m[8], l[8];
#pragma unroll
  for (int i = 0; i < 8; ++i) { m[i] = -1e30f; l[i] = 0.0f; }

  const float scale = 0.125f;          // 1/sqrt(64)

  for (int kt = 0; kt <= qblock; ++kt) {
    const int keyBase = kt * 128;

    // Warm L2/WGP$ for the next key tile (global_prefetch_b8).
    if (kt + 1 <= qblock) {
#pragma unroll
      for (int j = 0; j < 4; ++j) {
        int prow = keyBase + 128 + lane + 32 * j;
        __builtin_prefetch(&Kp[(size_t)prow * RS], 0, 1);
        __builtin_prefetch(&Vp[(size_t)prow * RS], 0, 1);
      }
    }

    __syncthreads();                   // protect Vt against overwrite
    // Stage V tile transposed: Vt[d][key], 128 keys x 64 d.
#pragma unroll
    for (int it = 0; it < 4; ++it) {
      int u = tid + it * 256;          // 0..1023
      int key = u >> 3;                // 8 uint4 per 64-wide d row
      int d = (u & 7) << 3;
      union { uint4 q; bf16_t h[8]; } t;
      t.q = *(const uint4*)&Vp[(size_t)(keyBase + key) * RS + d];
#pragma unroll
      for (int j = 0; j < 8; ++j) Vt[d + j][key] = t.h[j];
    }
    __syncthreads();

    // S[16 x 128] = Q_strip * K^T (8 n-tiles, 2 k-steps over HD=64)
    f32x8 s[8];
#pragma unroll
    for (int nt = 0; nt < 8; ++nt) s[nt] = zed;
#pragma unroll
    for (int kstep = 0; kstep < 2; ++kstep) {
#pragma unroll
      for (int nt = 0; nt < 8; ++nt) {
        FragU kb;
        int col = keyBase + nt * 16 + r;     // key index = B column
        int koff = kstep * 32 + g * 16;      // d range for this lane group
        kb.q[0] = *(const uint4*)&Kp[(size_t)col * RS + koff];
        kb.q[1] = *(const uint4*)&Kp[(size_t)col * RS + koff + 8];
        s[nt] = __builtin_amdgcn_wmma_f32_16x16x32_bf16(
            false, qa[kstep].v, false, kb.v, (short)0, s[nt], false, false);
      }
    }

    // Scale + causal mask (only the diagonal tile needs per-element masking).
    const bool diag = (kt == qblock);
#pragma unroll
    for (int nt = 0; nt < 8; ++nt)
#pragma unroll
      for (int i = 0; i < 8; ++i) {
        float v = s[nt][i] * scale;
        if (diag) {
          int qrow = q0 + i + g * 8;
          int key = keyBase + nt * 16 + r;
          v = (key <= qrow) ? v : -1e30f;
        }
        s[nt][i] = v;
      }

    // Online softmax. Row i+8g is replicated across the 16-lane group, so
    // row reductions are shfl_xor over masks 1,2,4,8 (stays within the group).
#pragma unroll
    for (int i = 0; i < 8; ++i) {
      float mx = s[0][i];
#pragma unroll
      for (int nt = 1; nt < 8; ++nt) mx = fmaxf(mx, s[nt][i]);
      mx = fmaxf(mx, __shfl_xor(mx, 1, 32));
      mx = fmaxf(mx, __shfl_xor(mx, 2, 32));
      mx = fmaxf(mx, __shfl_xor(mx, 4, 32));
      mx = fmaxf(mx, __shfl_xor(mx, 8, 32));
      float mnew = fmaxf(m[i], mx);
      float fac = __expf(m[i] - mnew);
      float rs = 0.0f;
#pragma unroll
      for (int nt = 0; nt < 8; ++nt) {
        float p = __expf(s[nt][i] - mnew);
        s[nt][i] = p;                  // reuse S registers as P
        rs += p;
      }
      rs += __shfl_xor(rs, 1, 32);
      rs += __shfl_xor(rs, 2, 32);
      rs += __shfl_xor(rs, 4, 32);
      rs += __shfl_xor(rs, 8, 32);
      l[i] = l[i] * fac + rs;
      m[i] = mnew;
#pragma unroll
      for (int vt = 0; vt < 4; ++vt) o[vt][i] *= fac;
    }

    // P: accumulator layout -> per-wave LDS strip (row-major 16x128 bf16).
#pragma unroll
    for (int nt = 0; nt < 8; ++nt)
#pragma unroll
      for (int i = 0; i < 8; ++i)
        Ps[wave][i + g * 8][nt * 16 + r] = f32_to_bf16(s[nt][i]);

    // O += P * V : 4 k-steps of 32 over the 128 keys, 4 d-tiles.
    // Same-wave LDS RAW on Ps: DS ops are in-order per wave; compiler waits.
#pragma unroll
    for (int kstep = 0; kstep < 4; ++kstep) {
      FragU pa;
      pa.q[0] = *(const uint4*)&Ps[wave][r][kstep * 32 + g * 16];
      pa.q[1] = *(const uint4*)&Ps[wave][r][kstep * 32 + g * 16 + 8];
#pragma unroll
      for (int vt = 0; vt < 4; ++vt) {
        FragU vb;
        int dcol = vt * 16 + r;
        int koff = kstep * 32 + g * 16;
        vb.q[0] = *(const uint4*)&Vt[dcol][koff];
        vb.q[1] = *(const uint4*)&Vt[dcol][koff + 8];
        o[vt] = __builtin_amdgcn_wmma_f32_16x16x32_bf16(
            false, pa.v, false, vb.v, (short)0, o[vt], false, false);
      }
    }
  }

  // Normalize and store bf16 attention output in [B,T,H*HD] layout.
#pragma unroll
  for (int i = 0; i < 8; ++i) {
    float inv = 1.0f / l[i];
    int trow = q0 + i + g * 8;
#pragma unroll
    for (int vt = 0; vt < 4; ++vt) {
      int d = vt * 16 + r;
      out[((size_t)bb * T_DIM + trow) * C_DIM + (size_t)hh * HD_DIM + d] =
          f32_to_bf16(o[vt][i] * inv);
    }
  }
}

// ---------------------------------------------------------------------------
// Launch
// ---------------------------------------------------------------------------
extern "C" void kernel_launch(void* const* d_in, const int* in_sizes, int n_in,
                              void* d_out, int out_size, void* d_ws, size_t ws_size,
                              hipStream_t stream) {
  const float* x      = (const float*)d_in[0];
  const float* w_qkv  = (const float*)d_in[1];
  const float* w_proj = (const float*)d_in[2];

  const size_t nX     = (size_t)B_DIM * T_DIM * C_DIM;       // 8,388,608
  const size_t nWqkv  = (size_t)C_DIM * 3 * C_DIM;           // 3,145,728
  const size_t nWproj = (size_t)C_DIM * C_DIM;               // 1,048,576
  const size_t nQKV   = (size_t)B_DIM * T_DIM * 3 * C_DIM;   // 25,165,824

  bf16_t* xb     = (bf16_t*)d_ws;
  bf16_t* wqkvb  = xb + nX;
  bf16_t* wprojb = wqkvb + nWqkv;
  bf16_t* qkvb   = wprojb + nWproj;
  bf16_t* attnb  = qkvb + nQKV;        // total ~92 MB of workspace

  convert_f32_bf16<<<2048, 256, 0, stream>>>(x, xb, (int)nX);
  convert_f32_bf16<<<2048, 256, 0, stream>>>(w_qkv, wqkvb, (int)nWqkv);
  convert_f32_bf16<<<1024, 256, 0, stream>>>(w_proj, wprojb, (int)nWproj);

  // qkv[B*T,3C] = x[B*T,C] @ w_qkv[C,3C]
  gemm_wmma<true><<<dim3(3 * C_DIM / 128, (B_DIM * T_DIM) / 128), 256, 0, stream>>>(
      xb, wqkvb, (void*)qkvb, B_DIM * T_DIM, 3 * C_DIM, C_DIM);

  // causal attention per (b,h)
  flash_attn<<<dim3(T_DIM / 128, B_DIM * H_DIM), 256, 0, stream>>>(qkvb, attnb);

  // out[B*T,C] = attn[B*T,C] @ w_proj[C,C], fp32 output
  gemm_wmma<false><<<dim3(C_DIM / 128, (B_DIM * T_DIM) / 128), 256, 0, stream>>>(
      attnb, wprojb, d_out, B_DIM * T_DIM, C_DIM, C_DIM);

  (void)in_sizes; (void)n_in; (void)out_size; (void)ws_size;
}
